// LlamaAttention_11716670784175
// MI455X (gfx1250) — matrix operand order repaired and checked by reading, not compiled, not run on hardware
//
#include <hip/hip_runtime.h>

#define Bc 2
#define Ec 1024
#define Dc 1024
#define Hc 2048
#define NHc 16
#define HDc 128
#define Sc 2048
#define NEGV (-1000000000.0f)
#define SCALEV 0.08838834764831845f  // 128^-0.5

typedef __attribute__((ext_vector_type(16))) __bf16 v16bf;
typedef __attribute__((ext_vector_type(8)))  __bf16 v8bf;
typedef __attribute__((ext_vector_type(8)))  float  v8f;

// ---------- helpers ----------

__device__ __forceinline__ unsigned short f2bf(float f) {
  unsigned int u = __builtin_bit_cast(unsigned int, f);
  u += 0x7FFFu + ((u >> 16) & 1u);           // round-to-nearest-even
  return (unsigned short)(u >> 16);
}

// A-fragment (16x32 bf16, row-major source, row stride ld).
// ISA 7.12.2: lane L -> row (L&15); VGPR0-3 hold K = 8*(L>=16)..+7, VGPR4-7 hold +16.
__device__ __forceinline__ v16bf frag_a_bf16(const __bf16* base, int ld, int m0, int k0, int lane) {
  const __bf16* p = base + (size_t)(m0 + (lane & 15)) * ld + k0 + ((lane >> 4) ? 8 : 0);
  v8bf lo = *(const v8bf*)p;          // 16B contiguous
  v8bf hi = *(const v8bf*)(p + 16);   // 16B contiguous
  v16bf a;
#pragma unroll
  for (int i = 0; i < 8; ++i) { a[i] = lo[i]; a[i + 8] = hi[i]; }
  return a;
}

// A-fragment from f32 source, converted to bf16 in registers (for attn @ v).
__device__ __forceinline__ v16bf frag_a_f32(const float* base, int ld, int m0, int k0, int lane) {
  const float* p = base + (size_t)(m0 + (lane & 15)) * ld + k0 + ((lane >> 4) ? 8 : 0);
  v16bf a;
#pragma unroll
  for (int i = 0; i < 8; ++i) {
    a[i]     = __builtin_bit_cast(__bf16, f2bf(p[i]));
    a[i + 8] = __builtin_bit_cast(__bf16, f2bf(p[i + 16]));
  }
  return a;
}

// B-fragment (32x16 bf16). B[k][n] = src[n][k], src row-major with row stride ld:
// lane (col n = L&15) reads K = k0 + 8*(L>=16)..+7 and +16 -> two 16B loads (matches A).
__device__ __forceinline__ v16bf frag_b_bf16(const __bf16* base, int ld, int k0, int n0, int lane) {
  const __bf16* p = base + (size_t)(n0 + (lane & 15)) * ld + k0 + ((lane >> 4) ? 8 : 0);
  v8bf lo = *(const v8bf*)p;
  v8bf hi = *(const v8bf*)(p + 16);
  v16bf b;
#pragma unroll
  for (int i = 0; i < 8; ++i) { b[i] = lo[i]; b[i + 8] = hi[i]; }
  return b;
}

__device__ __forceinline__ v8f wmma_bf16(v16bf a, v16bf b, v8f c) {
#if defined(__gfx1250__)
  return __builtin_amdgcn_wmma_f32_16x16x32_bf16(false, a, false, b, (short)0, c, false, false);
#else
  return c;
#endif
}

// Double-buffered 16x64 tile GEMM: acc[0..3] += A(16xK) * B(Kx64), K multiple of 32.
// Operand fragments for step k+1 are issued before the WMMAs of step k so the
// scheduler overlaps global_load_b128 with v_wmma instead of s_wait_loadcnt 0.
template <bool AF32>
__device__ __forceinline__ void gemm_16x64(const void* Abase, int lda,
                                           const __bf16* Bbase, int ldb,
                                           int m0, int n0, int K, int lane,
                                           v8f acc[4]) {
  v16bf a_cur, b_cur[4], a_nxt, b_nxt[4];
  if constexpr (AF32) a_cur = frag_a_f32((const float*)Abase, lda, m0, 0, lane);
  else                a_cur = frag_a_bf16((const __bf16*)Abase, lda, m0, 0, lane);
#pragma unroll
  for (int j = 0; j < 4; ++j) b_cur[j] = frag_b_bf16(Bbase, ldb, 0, n0 + 16 * j, lane);

  for (int k = 0; k < K; k += 32) {
    int kn = (k + 32 < K) ? (k + 32) : k;   // last iter: harmless reload, uniform flow
    if constexpr (AF32) {
      a_nxt = frag_a_f32((const float*)Abase, lda, m0, kn, lane);
      __builtin_prefetch((const void*)((const float*)Abase +
                         (size_t)(m0 + (lane & 15)) * lda + kn + 32), 0, 1);
    } else {
      a_nxt = frag_a_bf16((const __bf16*)Abase, lda, m0, kn, lane);
    }
#pragma unroll
    for (int j = 0; j < 4; ++j) b_nxt[j] = frag_b_bf16(Bbase, ldb, kn, n0 + 16 * j, lane);
#pragma unroll
    for (int j = 0; j < 4; ++j) acc[j] = wmma_bf16(a_cur, b_cur[j], acc[j]);
    a_cur = a_nxt;
#pragma unroll
    for (int j = 0; j < 4; ++j) b_cur[j] = b_nxt[j];
  }
}

// ---------- elementwise kernels ----------

__global__ void k_cvt_bf16(const float* __restrict__ src, unsigned short* __restrict__ dst, int n) {
  int i = blockIdx.x * 256 + threadIdx.x;
  if (i < n) dst[i] = f2bf(src[i]);
}

__global__ void k_pack_hs(const float* __restrict__ enc, const float* __restrict__ dec,
                          unsigned short* __restrict__ hsb) {
  int i = blockIdx.x * 256 + threadIdx.x;               // over B*S*H = 8388608
  int h = i & (Hc - 1);
  int t = i >> 11;                                      // /H
  int s = t & (Sc - 1);
  int b = t >> 11;                                      // /S
  float v = (s < Ec) ? enc[((size_t)b * Ec + s) * Hc + h]
                     : dec[((size_t)b * Dc + (s - Ec)) * Hc + h];
  hsb[i] = f2bf(v);
}

// RoPE on decoder rows of q/k, pack q,k -> bf16 (b,h,s,hd), v -> bf16 transposed (b,h,hd,s)
__global__ void k_rope_pack(const float* __restrict__ qf, const float* __restrict__ kf,
                            const float* __restrict__ vf,
                            const float* __restrict__ cosp, const float* __restrict__ sinp,
                            unsigned short* __restrict__ qb, unsigned short* __restrict__ kb,
                            unsigned short* __restrict__ vT) {
  size_t i = (size_t)blockIdx.x * 256 + threadIdx.x;    // over B*NH*S*HD = 8388608
  int hd = (int)(i & (HDc - 1));
  size_t t = i >> 7;
  int s = (int)(t & (Sc - 1));  t >>= 11;
  int h = (int)(t & (NHc - 1));
  int b = (int)(t >> 4);
  float q = qf[i], k = kf[i], v = vf[i];
  if (s >= Ec) {
    int p = s - Ec;
    float c  = cosp[((size_t)b * Dc + p) * HDc + hd];
    float si = sinp[((size_t)b * Dc + p) * HDc + hd];
    size_t pair = (hd < 64) ? (i + 64) : (i - 64);
    float qp = qf[pair], kp = kf[pair];
    float qr = (hd < 64) ? -qp : qp;
    float kr = (hd < 64) ? -kp : kp;
    q = q * c + qr * si;
    k = k * c + kr * si;
  }
  qb[i] = f2bf(q);
  kb[i] = f2bf(k);
  vT[(((size_t)b * NHc + h) * HDc + hd) * Sc + s] = f2bf(v);
}

// in-place row softmax over attn rows of length S
__global__ void k_softmax(float* __restrict__ attn) {
  __shared__ float red[256];
  float* p = attn + (size_t)blockIdx.x * Sc;            // B*NH*S rows
  int tid = threadIdx.x;
  float vals[8];
  float vmax = -3.4e38f;
#pragma unroll
  for (int i = 0; i < 8; ++i) { vals[i] = p[i * 256 + tid]; vmax = fmaxf(vmax, vals[i]); }
  red[tid] = vmax; __syncthreads();
  for (int o = 128; o > 0; o >>= 1) { if (tid < o) red[tid] = fmaxf(red[tid], red[tid + o]); __syncthreads(); }
  float m = red[0]; __syncthreads();
  float sum = 0.f;
#pragma unroll
  for (int i = 0; i < 8; ++i) { vals[i] = __expf(vals[i] - m); sum += vals[i]; }
  red[tid] = sum; __syncthreads();
  for (int o = 128; o > 0; o >>= 1) { if (tid < o) red[tid] += red[tid + o]; __syncthreads(); }
  float inv = 1.0f / red[0];
#pragma unroll
  for (int i = 0; i < 8; ++i) p[i * 256 + tid] = vals[i] * inv;
}

// ---------- WMMA GEMM kernels (each wave: 16x64 output strip) ----------

// qkv projection: dst[b,head,s,hd] = sum_h hs[b,s,h] * W[o=head*128+hd, h]
__global__ void k_proj_gemm(const unsigned short* __restrict__ hsb,
                            const unsigned short* __restrict__ wb,
                            float* __restrict__ dst) {
  int wave = (blockIdx.x * blockDim.x + threadIdx.x) >> 5;
  int lane = threadIdx.x & 31;
  const int NT = Hc / 64, MT = Sc / 16;                 // 32, 128
  int ntg = wave % NT; int t = wave / NT;
  int mt = t % MT;     int b = t / MT;                  // b < 2 exact
  int m0 = mt * 16, n0 = ntg * 64;
  const __bf16* A  = (const __bf16*)hsb + (size_t)b * Sc * Hc;
  const __bf16* Bw = (const __bf16*)wb;
  v8f acc[4] = {};
  gemm_16x64<false>(A, Hc, Bw, Hc, m0, n0, Hc, lane, acc);
  int rbase = (lane >> 4) * 8, nl = lane & 15;
#pragma unroll
  for (int j = 0; j < 4; ++j) {
    int n = n0 + 16 * j + nl;
    size_t head = (size_t)(n >> 7); int hd = n & 127;
#pragma unroll
    for (int r = 0; r < 8; ++r) {
      int m = m0 + rbase + r;
      dst[(((size_t)b * NHc + head) * Sc + m) * HDc + hd] = acc[j][r];
    }
  }
}

// scores[b,h,m,n] = scale * (q[m,:] . k[n,:]) + mask
__global__ void k_scores_gemm(const unsigned short* __restrict__ qb,
                              const unsigned short* __restrict__ kb,
                              const float* __restrict__ amask,
                              float* __restrict__ attn) {
  int wave = (blockIdx.x * blockDim.x + threadIdx.x) >> 5;
  int lane = threadIdx.x & 31;
  const int NT = Sc / 64, MT = Sc / 16;                 // 32, 128
  int ntg = wave % NT; int t = wave / NT;
  int mt = t % MT;     int bh = t / MT;                 // bh < 32 exact
  int m0 = mt * 16, n0 = ntg * 64;
  const __bf16* Q = (const __bf16*)qb + (size_t)bh * Sc * HDc;
  const __bf16* K = (const __bf16*)kb + (size_t)bh * Sc * HDc;
  v8f acc[4] = {};
  gemm_16x64<false>(Q, HDc, K, HDc, m0, n0, HDc, lane, acc);
  int b = bh >> 4;
  int rbase = (lane >> 4) * 8, nl = lane & 15;
#pragma unroll
  for (int j = 0; j < 4; ++j) {
    int n = n0 + 16 * j + nl;
#pragma unroll
    for (int r = 0; r < 8; ++r) {
      int m = m0 + rbase + r;
      float mk;
      if (m < Ec) mk = (n < Ec) ? 0.f : NEGV;
      else        mk = (n < Ec) ? 0.f
                                : amask[((size_t)b * Dc + (m - Ec)) * Dc + (n - Ec)];
      attn[((size_t)bh * Sc + m) * Sc + n] = acc[j][r] * SCALEV + mk;
    }
  }
}

// ctx[b, m(dec), head*128+n] = sum_s attn[b,h,E+m,s] * v[b,h,s,n]   (v given transposed)
__global__ void k_av_gemm(const float* __restrict__ attn,
                          const unsigned short* __restrict__ vT,
                          unsigned short* __restrict__ ctx) {
  int wave = (blockIdx.x * blockDim.x + threadIdx.x) >> 5;
  int lane = threadIdx.x & 31;
  const int NT = HDc / 64, MT = Dc / 16;                // 2, 64
  int ntg = wave % NT; int t = wave / NT;
  int mt = t % MT;     int bh = t / MT;                 // bh < 32 exact
  int m0 = mt * 16, n0 = ntg * 64;
  const float*  A  = attn + (size_t)bh * Sc * Sc + (size_t)Ec * Sc;  // decoder rows
  const __bf16* Bv = (const __bf16*)vT + (size_t)bh * HDc * Sc;
  v8f acc[4] = {};
  gemm_16x64<true>(A, Sc, Bv, Sc, m0, n0, Sc, lane, acc);
  int b = bh >> 4, h = bh & 15;
  int rbase = (lane >> 4) * 8, nl = lane & 15;
#pragma unroll
  for (int j = 0; j < 4; ++j) {
    int n = n0 + 16 * j + nl;
#pragma unroll
    for (int r = 0; r < 8; ++r) {
      int m = m0 + rbase + r;
      ctx[((size_t)b * Dc + m) * Hc + h * HDc + n] = f2bf(acc[j][r]);
    }
  }
}

// out[b,m,n] = sum_k ctx[b,m,k] * Wo[n,k]
__global__ void k_out_gemm(const unsigned short* __restrict__ ctx,
                           const unsigned short* __restrict__ wob,
                           float* __restrict__ out) {
  int wave = (blockIdx.x * blockDim.x + threadIdx.x) >> 5;
  int lane = threadIdx.x & 31;
  const int NT = Hc / 64, MT = Dc / 16;                 // 32, 64
  int ntg = wave % NT; int t = wave / NT;
  int mt = t % MT;     int b = t / MT;                  // b < 2 exact
  int m0 = mt * 16, n0 = ntg * 64;
  const __bf16* A  = (const __bf16*)ctx + (size_t)b * Dc * Hc;
  const __bf16* Bw = (const __bf16*)wob;
  v8f acc[4] = {};
  gemm_16x64<false>(A, Hc, Bw, Hc, m0, n0, Hc, lane, acc);
  int rbase = (lane >> 4) * 8, nl = lane & 15;
#pragma unroll
  for (int j = 0; j < 4; ++j) {
    int n = n0 + 16 * j + nl;
#pragma unroll
    for (int r = 0; r < 8; ++r) {
      int m = m0 + rbase + r;
      out[((size_t)b * Dc + m) * Hc + n] = acc[j][r];
    }
  }
}

// ---------- launcher ----------

extern "C" void kernel_launch(void* const* d_in, const int* in_sizes, int n_in,
                              void* d_out, int out_size, void* d_ws, size_t ws_size,
                              hipStream_t stream) {
  const float* hidden = (const float*)d_in[0];
  const float* enc    = (const float*)d_in[1];
  const float* cosp   = (const float*)d_in[2];
  const float* sinp   = (const float*)d_in[3];
  const float* amask  = (const float*)d_in[4];
  const float* Wq     = (const float*)d_in[5];
  const float* Wk     = (const float*)d_in[6];
  const float* Wv     = (const float*)d_in[7];
  const float* Wo     = (const float*)d_in[8];

  float* out  = (float*)d_out;                                  // (B,D,H)
  float* attn = out + (size_t)Bc * Dc * Hc;                     // (B,NH,S,S) in d_out

  // workspace layout (256B aligned bumps)
  char* w = (char*)d_ws;
  auto bump = [&](size_t bytes) { char* p = w; w += (bytes + 255) & ~(size_t)255; return p; };
  unsigned short* hsb = (unsigned short*)bump((size_t)Bc * Sc * Hc * 2);
  unsigned short* wqb = (unsigned short*)bump((size_t)Hc * Hc * 2);
  unsigned short* wkb = (unsigned short*)bump((size_t)Hc * Hc * 2);
  unsigned short* wvb = (unsigned short*)bump((size_t)Hc * Hc * 2);
  unsigned short* wob = (unsigned short*)bump((size_t)Hc * Hc * 2);
  float* qf = (float*)bump((size_t)Bc * NHc * Sc * HDc * 4);
  float* kf = (float*)bump((size_t)Bc * NHc * Sc * HDc * 4);
  float* vf = (float*)bump((size_t)Bc * NHc * Sc * HDc * 4);
  unsigned short* qb2 = (unsigned short*)bump((size_t)Bc * NHc * Sc * HDc * 2);
  unsigned short* kb2 = (unsigned short*)bump((size_t)Bc * NHc * Sc * HDc * 2);
  unsigned short* vT  = (unsigned short*)bump((size_t)Bc * NHc * HDc * Sc * 2);
  unsigned short* ctx = (unsigned short*)bump((size_t)Bc * Dc * Hc * 2);

  const int nW = Hc * Hc;                 // 4194304
  const int nHS = Bc * Sc * Hc;           // 8388608
  const int nQKV = Bc * NHc * Sc * HDc;   // 8388608

  // 1) pack + convert to bf16
  k_pack_hs<<<nHS / 256, 256, 0, stream>>>(enc, hidden, hsb);
  k_cvt_bf16<<<nW / 256, 256, 0, stream>>>(Wq, wqb, nW);
  k_cvt_bf16<<<nW / 256, 256, 0, stream>>>(Wk, wkb, nW);
  k_cvt_bf16<<<nW / 256, 256, 0, stream>>>(Wv, wvb, nW);
  k_cvt_bf16<<<nW / 256, 256, 0, stream>>>(Wo, wob, nW);

  // 2) q/k/v projections: waves = B*(S/16)*(H/64) = 8192 -> 1024 blocks of 8 waves
  k_proj_gemm<<<1024, 256, 0, stream>>>(hsb, wqb, qf);
  k_proj_gemm<<<1024, 256, 0, stream>>>(hsb, wkb, kf);
  k_proj_gemm<<<1024, 256, 0, stream>>>(hsb, wvb, vf);

  // 3) RoPE + pack to bf16 (+ v transpose)
  k_rope_pack<<<nQKV / 256, 256, 0, stream>>>(qf, kf, vf, cosp, sinp, qb2, kb2, vT);

  // 4) scores: waves = 32*(S/16)*(S/64) = 131072 -> 16384 blocks
  k_scores_gemm<<<16384, 256, 0, stream>>>(qb2, kb2, amask, attn);

  // 5) softmax in place on attn (rows = B*NH*S = 65536)
  k_softmax<<<Bc * NHc * Sc, 256, 0, stream>>>(attn);

  // 6) attn @ v (decoder rows only): waves = 32*(D/16)*(HD/64) = 4096 -> 512 blocks
  k_av_gemm<<<512, 256, 0, stream>>>(attn, vT, ctx);

  // 7) output projection: waves = B*(D/16)*(H/64) = 4096 -> 512 blocks
  k_out_gemm<<<512, 256, 0, stream>>>(ctx, wob, out);
}
